// PairEnergies_50646254354670
// MI455X (gfx1250) — matrix operand order repaired and checked by reading, not compiled, not run on hardware
//
#include <hip/hip_runtime.h>

// ---------------------------------------------------------------------------
// PairEnergies forward for MI455X (gfx1250, wave32, WMMA).
// All GEMMs run through v_wmma_f32_16x16x32_f16 (f16 in, f32 accumulate).
// ---------------------------------------------------------------------------

typedef __attribute__((ext_vector_type(16))) _Float16 v16h;
typedef __attribute__((ext_vector_type(8)))  float    v8f;

constexpr int Bc   = 2;
constexpr int Nc   = 768;
constexpr int Kc   = 30;
constexpr int HD   = 128;
constexpr int DINc = 128;
constexpr int OD   = 400;
constexpr int NLAY = 3;
constexpr int NV   = Bc * Nc;        // 1536 nodes
constexpr int NE   = Bc * Nc * Kc;   // 46080 edges

// ---------------------------------------------------------------------------
// Generic f16 WMMA GEMM:  C[M x Nd] = act(A[M x Kd] * W[Kd x Nd] + bias)
// A: row-major f16.  Wt: W transposed, row-major [Nd x Kd] f16 (so each lane's
// B fragment is 16 contiguous halves = one 32B load).
// One wave -> one 16x16 C tile.  M % 16 == 0, Kd % 32 == 0 (always true here).
// ---------------------------------------------------------------------------
template <bool RELU, bool WF32, bool WF16>
__global__ __launch_bounds__(128)
void gemm_f16_wmma(const _Float16* __restrict__ A,
                   const _Float16* __restrict__ Wt,
                   const float* __restrict__ bias,
                   float* __restrict__ Cf, _Float16* __restrict__ Ch,
                   int M, int Kd, int Nd)
{
    const int lane = threadIdx.x & 31;
    const int wave = threadIdx.x >> 5;
    const int tm   = blockIdx.x;
    const int tn   = blockIdx.y * 4 + wave;
    if (tn * 16 >= Nd) return;                 // wave-uniform; EXEC stays ~0 lanes off
    const int hi  = lane >> 4;                 // 0 or 1 (lane half)
    const int l16 = lane & 15;
    const int m   = tm * 16 + l16;             // A row held by this lane
    const int n   = tn * 16 + l16;             // C column / Wt row
    const _Float16* Arow = A  + (size_t)m * Kd;
    const _Float16* Wrow = Wt + (size_t)n * Kd;

    v8f acc = {};
    for (int k0 = 0; k0 < Kd; k0 += 32) {
        // A 16x32 f16 layout: lane half hi, VGPR v holds K = kb(v), kb(v)+1
        v16h a;
#pragma unroll
        for (int v = 0; v < 8; ++v) {
            const int kb = k0 + 2 * v + 16 * hi;
            a[2 * v]     = Arow[kb];
            a[2 * v + 1] = Arow[kb + 1];
        }
        // B 32x16: lanes 0-15 hold K=k0..k0+15 of column n, lanes 16-31 K=+16
        v16h b = *(const v16h*)(Wrow + k0 + 16 * hi);
        acc = __builtin_amdgcn_wmma_f32_16x16x32_f16(false, a, false, b,
                                                     (short)0, acc, false, false);
    }
    const float bb = bias ? bias[n] : 0.f;
#pragma unroll
    for (int r = 0; r < 8; ++r) {
        const int mr = tm * 16 + r + 8 * hi;   // C/D layout: VGPR r -> M = r + 8*hi
        float v = acc[r] + bb;
        if (RELU) v = v > 0.f ? v : 0.f;
        if (WF32) Cf[(size_t)mr * Nd + n] = v;
        if (WF16) Ch[(size_t)mr * Nd + n] = (_Float16)v;
    }
}

// ---------------------------------------------------------------------------
// Weight fp32 [Kd x Nd] -> f16 transposed [Nd x Kd]
// ---------------------------------------------------------------------------
__global__ void wt_transpose(const float* __restrict__ W, _Float16* __restrict__ T,
                             int Kd, int Nd)
{
    int id = blockIdx.x * blockDim.x + threadIdx.x;
    if (id >= Kd * Nd) return;
    int k = id / Nd, n = id % Nd;
    T[(size_t)n * Kd + k] = (_Float16)W[id];
}

// ---------------------------------------------------------------------------
// kNN: per (b,i) find K nearest (masked) CA neighbors, stable tie-break by idx.
// ---------------------------------------------------------------------------
__global__ __launch_bounds__(256)
void knn_kernel(const float* __restrict__ X, const float* __restrict__ xm,
                int* __restrict__ Eidx, float* __restrict__ Dnbr)
{
    const int b = blockIdx.x / Nc, i = blockIdx.x % Nc, t = threadIdx.x;
    __shared__ float sD[Nc];
    __shared__ float sA[Nc];
    __shared__ float rv[256];
    __shared__ int   ri[256];
    const float* Xi = X + (((size_t)(b * Nc + i)) * 4 + 1) * 3;
    const float xi0 = Xi[0], xi1 = Xi[1], xi2 = Xi[2];
    const float mi = xm[b * Nc + i];
    for (int j = t; j < Nc; j += 256) {
        const float* Xj = X + (((size_t)(b * Nc + j)) * 4 + 1) * 3;
        float dx = xi0 - Xj[0], dy = xi1 - Xj[1], dz = xi2 - Xj[2];
        float D = sqrtf(dx * dx + dy * dy + dz * dz + 1e-6f);
        sD[j] = D;
        sA[j] = (mi * xm[b * Nc + j] > 0.f) ? D : D + 1e6f;
    }
    __syncthreads();
    for (int k = 0; k < Kc; ++k) {
        float bv = 3.0e38f; int bi = 0x7fffffff;
        for (int j = t; j < Nc; j += 256) {
            float v = sA[j];
            if (v < bv || (v == bv && j < bi)) { bv = v; bi = j; }
        }
        rv[t] = bv; ri[t] = bi; __syncthreads();
        for (int s = 128; s > 0; s >>= 1) {
            if (t < s) {
                if (rv[t + s] < rv[t] || (rv[t + s] == rv[t] && ri[t + s] < ri[t])) {
                    rv[t] = rv[t + s]; ri[t] = ri[t + s];
                }
            }
            __syncthreads();
        }
        if (t == 0) {
            int j = ri[0];
            Eidx[(size_t)blockIdx.x * Kc + k] = j;
            Dnbr[(size_t)blockIdx.x * Kc + k] = sD[j];
            sA[j] = 3.0e38f;
        }
        __syncthreads();
    }
}

// ---------------------------------------------------------------------------
// Dihedral angles (3 per residue, padded as in reference).
// ---------------------------------------------------------------------------
__global__ void dihedral_kernel(const float* __restrict__ X, float* __restrict__ ang)
{
    int id = blockIdx.x * blockDim.x + threadIdx.x;
    if (id >= Bc * Nc * 3) return;
    const int b = id / (Nc * 3);
    const int j = id % (Nc * 3);
    const int L = 3 * Nc;
    float out = 0.f;
    if (j >= 1 && j <= L - 3) {
        const int t0 = j - 1;
        float P[4][3];
#pragma unroll
        for (int s = 0; s < 4; ++s) {
            int i = t0 + s, r = i / 3, a = i % 3;
            const float* p = X + (((size_t)(b * Nc + r)) * 4 + a) * 3;
            P[s][0] = p[0]; P[s][1] = p[1]; P[s][2] = p[2];
        }
        float U[3][3];
#pragma unroll
        for (int s = 0; s < 3; ++s) {
            float dx = P[s + 1][0] - P[s][0], dy = P[s + 1][1] - P[s][1], dz = P[s + 1][2] - P[s][2];
            float nn = sqrtf(dx * dx + dy * dy + dz * dz) + 1e-7f;
            U[s][0] = dx / nn; U[s][1] = dy / nn; U[s][2] = dz / nn;
        }
        float c2[3] = { U[0][1] * U[1][2] - U[0][2] * U[1][1],
                        U[0][2] * U[1][0] - U[0][0] * U[1][2],
                        U[0][0] * U[1][1] - U[0][1] * U[1][0] };
        float c1[3] = { U[1][1] * U[2][2] - U[1][2] * U[2][1],
                        U[1][2] * U[2][0] - U[1][0] * U[2][2],
                        U[1][0] * U[2][1] - U[1][1] * U[2][0] };
        float n2n = sqrtf(c2[0] * c2[0] + c2[1] * c2[1] + c2[2] * c2[2]) + 1e-7f;
        float n1n = sqrtf(c1[0] * c1[0] + c1[1] * c1[1] + c1[2] * c1[2]) + 1e-7f;
        float cosD = (c2[0] * c1[0] + c2[1] * c1[1] + c2[2] * c1[2]) / (n2n * n1n);
        cosD = fminf(fmaxf(cosD, -1.f + 1e-7f), 1.f - 1e-7f);
        float sg = (U[0][0] * c1[0] + U[0][1] * c1[1] + U[0][2] * c1[2]) / n1n;
        float s = (sg > 0.f) ? 1.f : ((sg < 0.f) ? -1.f : 0.f);
        out = s * acosf(cosD);
    }
    ang[id] = out;
}

// ---------------------------------------------------------------------------
// V features: LN(dih6 @ Wvf + bvf) -> left half of A_hv; V_embed -> right half.
// ---------------------------------------------------------------------------
__global__ __launch_bounds__(128)
void vfeat_kernel(const float* __restrict__ ang,
                  const float* __restrict__ Wvf, const float* __restrict__ bvf,
                  const float* __restrict__ gvf, const float* __restrict__ nvf,
                  const float* __restrict__ Vemb, _Float16* __restrict__ Ahv)
{
    const int row = blockIdx.x, t = threadIdx.x;
    __shared__ float f[6];
    __shared__ float red[128];
    __shared__ float mu_s, var_s;
    if (t < 3) { float a = ang[row * 3 + t]; f[t] = cosf(a); f[t + 3] = sinf(a); }
    __syncthreads();
    float v = bvf[t];
#pragma unroll
    for (int i = 0; i < 6; ++i) v += f[i] * Wvf[i * HD + t];
    red[t] = v; __syncthreads();
    for (int s = 64; s > 0; s >>= 1) { if (t < s) red[t] += red[t + s]; __syncthreads(); }
    if (t == 0) mu_s = red[0] / 128.f;
    __syncthreads();
    float d = v - mu_s;
    red[t] = d * d; __syncthreads();
    for (int s = 64; s > 0; s >>= 1) { if (t < s) red[t] += red[t + s]; __syncthreads(); }
    if (t == 0) var_s = red[0] / 128.f;
    __syncthreads();
    float o = gvf[t] * d * rsqrtf(var_s + 1e-5f) + nvf[t];
    Ahv[(size_t)row * 256 + t]       = (_Float16)o;
    Ahv[(size_t)row * 256 + 128 + t] = (_Float16)Vemb[(size_t)row * HD + t];
}

// ---------------------------------------------------------------------------
// Edge raw features (pos-embed 16 + RBF 16) and E_nbr gather into A_he[:,128:].
// ---------------------------------------------------------------------------
__global__ __launch_bounds__(128)
void eraw_kernel(const int* __restrict__ Eidx, const int* __restrict__ chain,
                 const float* __restrict__ Dnbr, const float* __restrict__ Eemb,
                 _Float16* __restrict__ Aeraw, _Float16* __restrict__ Ahe)
{
    const int e = blockIdx.x, t = threadIdx.x;
    const int b = e / (Nc * Kc), i = (e / Kc) % Nc;
    const int j = Eidx[e];
    if (t < 32) {
        float val;
        if (t < 16) {
            const int m = t & 7;
            float dd = (float)(j - i);
            if (chain[b * Nc + j] != chain[b * Nc + i]) dd = 0.f;
            float a = dd * expf(-1.1512925465f * (float)m);  // 2m*ln(1e4)/16
            val = (t < 8) ? cosf(a) : sinf(a);
        } else {
            const int m = t - 16;
            float mu = 2.f + (20.f / 15.f) * (float)m;       // linspace(2,22,16)
            float x  = (Dnbr[e] - mu) / 1.25f;               // sigma = 20/16
            val = expf(-x * x);
        }
        Aeraw[(size_t)e * 32 + t] = (_Float16)val;
    }
    Ahe[(size_t)e * 256 + 128 + t] =
        (_Float16)Eemb[((size_t)(b * Nc + i) * Nc + j) * DINc + t];
}

// ---------------------------------------------------------------------------
// Fused (X + premask*Y) -> LayerNorm(g,b) -> *postmask, optional f32/f16 out.
// C = 128, one block per row.
// ---------------------------------------------------------------------------
__global__ __launch_bounds__(128)
void fused_add_ln(const float* __restrict__ X, const float* __restrict__ Y,
                  const float* __restrict__ premask,
                  const float* __restrict__ g, const float* __restrict__ bt,
                  const float* __restrict__ postmask,
                  float* __restrict__ outf, _Float16* __restrict__ outh,
                  int ostrideH, int ooffH)
{
    const int row = blockIdx.x, t = threadIdx.x;
    __shared__ float red[128];
    __shared__ float mu_s, var_s;
    float v = X[(size_t)row * 128 + t];
    if (Y) {
        float y = Y[(size_t)row * 128 + t];
        if (premask) y *= premask[row];
        v += y;
    }
    red[t] = v; __syncthreads();
    for (int s = 64; s > 0; s >>= 1) { if (t < s) red[t] += red[t + s]; __syncthreads(); }
    if (t == 0) mu_s = red[0] / 128.f;
    __syncthreads();
    float d = v - mu_s;
    red[t] = d * d; __syncthreads();
    for (int s = 64; s > 0; s >>= 1) { if (t < s) red[t] += red[t + s]; __syncthreads(); }
    if (t == 0) var_s = red[0] / 128.f;
    __syncthreads();
    float o = g[t] * d * rsqrtf(var_s + 1e-5f) + bt[t];
    if (postmask) o *= postmask[row];
    if (outf) outf[(size_t)row * 128 + t] = o;
    if (outh) outh[(size_t)row * ostrideH + ooffH + t] = (_Float16)o;
}

// ---------------------------------------------------------------------------
// Edge masks, reverse-edge lookup, duplicate-merge.
// ---------------------------------------------------------------------------
__global__ void edge_masks(const int* __restrict__ Eidx, const float* __restrict__ xm,
                           float* __restrict__ mAtt, float* __restrict__ mE)
{
    int e = blockIdx.x * blockDim.x + threadIdx.x;
    if (e >= NE) return;
    int b = e / (Nc * Kc), i = (e / Kc) % Nc, j = Eidx[e];
    float mi = xm[b * Nc + i];
    mAtt[e] = mi * xm[b * Nc + j];
    mE[e]   = mi;
}

__global__ void reverse_edges(const int* __restrict__ Eidx,
                              int* __restrict__ ex, int* __restrict__ kp)
{
    int e = blockIdx.x * blockDim.x + threadIdx.x;
    if (e >= NE) return;
    int b = e / (Nc * Kc), i = (e / Kc) % Nc, j = Eidx[e];
    const int* row = Eidx + (size_t)(b * Nc + j) * Kc;
    int found = 0, kk0 = 0;
    for (int kk = 0; kk < Kc; ++kk)
        if (row[kk] == i) { found = 1; kk0 = kk; break; }
    ex[e] = found; kp[e] = kk0;
}

__global__ __launch_bounds__(128)
void merge128(const float* __restrict__ in, float* __restrict__ out,
              const int* __restrict__ Eidx, const int* __restrict__ ex,
              const int* __restrict__ kp)
{
    const int e = blockIdx.x, c = threadIdx.x;
    const int b = e / (Nc * Kc);
    float v = in[(size_t)e * HD + c];
    if (ex[e]) {
        size_t e2 = (size_t)(b * Nc + Eidx[e]) * Kc + kp[e];
        v = 0.5f * (v + in[e2 * HD + c]);
    }
    out[(size_t)e * HD + c] = v;
}

// ---------------------------------------------------------------------------
// Concat builders (f32 -> f16 GEMM inputs).
// ---------------------------------------------------------------------------
__global__ __launch_bounds__(512)
void build_edge_x(const float* __restrict__ hE, const float* __restrict__ hV,
                  const int* __restrict__ Eidx, _Float16* __restrict__ R1)
{
    const int e = blockIdx.x, c = threadIdx.x;   // 512 wide: [hE, hV_i, hV_j, hE]
    const int b = e / (Nc * Kc), i = (e / Kc) % Nc;
    float v;
    if (c < 128)      v = hE[(size_t)e * HD + c];
    else if (c < 256) v = hV[(size_t)(b * Nc + i) * HD + (c - 128)];
    else if (c < 384) v = hV[(size_t)(b * Nc + Eidx[e]) * HD + (c - 256)];
    else              v = hE[(size_t)e * HD + (c - 384)];
    R1[(size_t)e * 512 + c] = (_Float16)v;
}

__global__ __launch_bounds__(384)
void build_node_x(const float* __restrict__ hE, const float* __restrict__ hV,
                  const int* __restrict__ Eidx, _Float16* __restrict__ R1)
{
    const int e = blockIdx.x, c = threadIdx.x;   // 384 wide: [hV_i, hE, hV_j]
    const int b = e / (Nc * Kc), i = (e / Kc) % Nc;
    float v;
    if (c < 128)      v = hV[(size_t)(b * Nc + i) * HD + c];
    else if (c < 256) v = hE[(size_t)e * HD + (c - 128)];
    else              v = hV[(size_t)(b * Nc + Eidx[e]) * HD + (c - 256)];
    R1[(size_t)e * 384 + c] = (_Float16)v;
}

// Node message aggregation: dh[v,c] = sum_k mask*m3[v,k,c] / SCALE
__global__ void node_reduce(const float* __restrict__ m3, const float* __restrict__ mAtt,
                            float* __restrict__ dh)
{
    int id = blockIdx.x * blockDim.x + threadIdx.x;
    if (id >= NV * HD) return;
    int row = id >> 7, c = id & 127;
    float s = 0.f;
    for (int k = 0; k < Kc; ++k) {
        size_t e = (size_t)row * Kc + k;
        s += m3[e * HD + c] * mAtt[e];
    }
    dh[id] = s / 30.0f;
}

// ---------------------------------------------------------------------------
// Output head post-processing.
// ---------------------------------------------------------------------------
__global__ __launch_bounds__(400)
void selfslot_mask(float* __restrict__ A, const float* __restrict__ mE)
{
    const int e = blockIdx.x, c = threadIdx.x;
    float v = A[(size_t)e * OD + c] * mE[e];
    if ((e % Kc) == 0) { int p = c / 20, q = c % 20; if (p != q) v = 0.f; }
    A[(size_t)e * OD + c] = v;
}

__global__ __launch_bounds__(400)
void merge_pairE(const float* __restrict__ A, const int* __restrict__ Eidx,
                 const int* __restrict__ ex, const int* __restrict__ kp,
                 float* __restrict__ out)
{
    const int e = blockIdx.x, c = threadIdx.x;
    const int b = e / (Nc * Kc);
    float v = A[(size_t)e * OD + c];
    if (ex[e]) {
        size_t e2 = (size_t)(b * Nc + Eidx[e]) * Kc + kp[e];
        int p = c / 20, q = c % 20;
        v = 0.5f * (v + A[e2 * OD + q * 20 + p]);   // transpose of reverse edge
    }
    out[(size_t)e * OD + c] = v;
}

// ---------------------------------------------------------------------------
// Small utilities.
// ---------------------------------------------------------------------------
__global__ void cvt_f32_f16(const float* __restrict__ s, _Float16* __restrict__ d, int n)
{ int i = blockIdx.x * blockDim.x + threadIdx.x; if (i < n) d[i] = (_Float16)s[i]; }

__global__ void copy_f32k(const float* __restrict__ s, float* __restrict__ d, int n)
{ int i = blockIdx.x * blockDim.x + threadIdx.x; if (i < n) d[i] = s[i]; }

__global__ void copy_i32k(const int* __restrict__ s, int* __restrict__ d, int n)
{ int i = blockIdx.x * blockDim.x + threadIdx.x; if (i < n) d[i] = s[i]; }

// ---------------------------------------------------------------------------
// Host orchestration.
// Param pytree order (sorted dict keys): d_in[5]=We, 6=Wef, 7=Wout, 8=Wv,
// 9=Wvf, 10=be, 11=bef, 12=bout, 13=bv, 14=bvf, 15..56=edge_layers[0..2]
// {W1,W2,W3,Wd1,Wd2,b1,b2,b3,bd1,bd2,g1,g2,n1,n2}, 57=gef, 58=gvf, 59=nef,
// 60..101=node_layers (same 14-leaf order), 102=nvf.
// ---------------------------------------------------------------------------
extern "C" void kernel_launch(void* const* d_in, const int* in_sizes, int n_in,
                              void* d_out, int out_size, void* d_ws, size_t ws_size,
                              hipStream_t stream)
{
    (void)in_sizes; (void)n_in; (void)out_size; (void)ws_size;
    const float* V_embed = (const float*)d_in[0];
    const float* E_embed = (const float*)d_in[1];
    const float* Xin     = (const float*)d_in[2];
    const float* x_mask  = (const float*)d_in[3];
    const int*   chain   = (const int*)d_in[4];
    auto F = [&](int i) { return (const float*)d_in[i]; };

    char* base = (char*)d_ws;
    size_t off = 0;
    auto alloc = [&](size_t bytes) -> char* {
        char* p = base + off;
        off += (bytes + 255) & ~(size_t)255;
        return p;
    };

    // ---- transposed f16 weights ----
    _Float16* WefT  = (_Float16*)alloc((size_t)32 * 128 * 2);
    _Float16* WvT   = (_Float16*)alloc((size_t)256 * 128 * 2);
    _Float16* WeT   = (_Float16*)alloc((size_t)256 * 128 * 2);
    _Float16* WoutT = (_Float16*)alloc((size_t)128 * 400 * 2);
    _Float16 *eW1T[NLAY], *eW2T[NLAY], *eW3T[NLAY], *eWd1T[NLAY], *eWd2T[NLAY];
    _Float16 *nW1T[NLAY], *nW2T[NLAY], *nW3T[NLAY], *nWd1T[NLAY], *nWd2T[NLAY];
    for (int l = 0; l < NLAY; ++l) {
        eW1T[l]  = (_Float16*)alloc((size_t)512 * 128 * 2);
        eW2T[l]  = (_Float16*)alloc((size_t)128 * 128 * 2);
        eW3T[l]  = (_Float16*)alloc((size_t)128 * 128 * 2);
        eWd1T[l] = (_Float16*)alloc((size_t)128 * 512 * 2);
        eWd2T[l] = (_Float16*)alloc((size_t)512 * 128 * 2);
        nW1T[l]  = (_Float16*)alloc((size_t)384 * 128 * 2);
        nW2T[l]  = (_Float16*)alloc((size_t)128 * 128 * 2);
        nW3T[l]  = (_Float16*)alloc((size_t)128 * 128 * 2);
        nWd1T[l] = (_Float16*)alloc((size_t)128 * 512 * 2);
        nWd2T[l] = (_Float16*)alloc((size_t)512 * 128 * 2);
    }

    // ---- activations / scratch ----
    int*      Eidx   = (int*)alloc((size_t)NE * 4);
    float*    Dnbr   = (float*)alloc((size_t)NE * 4);
    int*      exst   = (int*)alloc((size_t)NE * 4);
    int*      kpr    = (int*)alloc((size_t)NE * 4);
    float*    angles = (float*)alloc((size_t)NV * 3 * 4);
    float*    mAtt   = (float*)alloc((size_t)NE * 4);
    float*    mE     = (float*)alloc((size_t)NE * 4);
    _Float16* Ahv    = (_Float16*)alloc((size_t)NV * 256 * 2);
    float*    hV     = (float*)alloc((size_t)NV * HD * 4);
    float*    hVln   = (float*)alloc((size_t)NV * HD * 4);
    float*    hVdh   = (float*)alloc((size_t)NV * HD * 4);
    _Float16* hVln16 = (_Float16*)alloc((size_t)NV * HD * 2);
    _Float16* hVd116 = (_Float16*)alloc((size_t)NV * 512 * 2);
    _Float16* Aeraw  = (_Float16*)alloc((size_t)NE * 32 * 2);
    float*    hE_a   = (float*)alloc((size_t)NE * HD * 4);
    float*    hE_b   = (float*)alloc((size_t)NE * HD * 4);
    float*    hEln   = (float*)alloc((size_t)NE * HD * 4);
    _Float16* hEln16 = (_Float16*)alloc((size_t)NE * HD * 2);
    _Float16* M1     = (_Float16*)alloc((size_t)NE * HD * 2);
    _Float16* M2     = (_Float16*)alloc((size_t)NE * HD * 2);
    float*    m3     = (float*)alloc((size_t)NE * HD * 4);
    _Float16* hE16   = (_Float16*)alloc((size_t)NE * HD * 2);
    _Float16* R1     = (_Float16*)alloc((size_t)NE * 512 * 2);
    float*    out400 = (float*)alloc((size_t)NE * OD * 4);
    _Float16* Ahe    = R1;   // [NE x 256] overlaps R1 (disjoint lifetimes)

    auto T = [&](const float* W, _Float16* Wt, int Kd, int Nd) {
        int n = Kd * Nd;
        wt_transpose<<<(n + 255) / 256, 256, 0, stream>>>(W, Wt, Kd, Nd);
    };
    T(F(6), WefT, 32, 128);
    T(F(8), WvT, 256, 128);
    T(F(5), WeT, 256, 128);
    T(F(7), WoutT, 128, 400);
    for (int l = 0; l < NLAY; ++l) {
        int be_l = 15 + 14 * l, bn_l = 60 + 14 * l;
        T(F(be_l + 0), eW1T[l], 512, 128);
        T(F(be_l + 1), eW2T[l], 128, 128);
        T(F(be_l + 2), eW3T[l], 128, 128);
        T(F(be_l + 3), eWd1T[l], 128, 512);
        T(F(be_l + 4), eWd2T[l], 512, 128);
        T(F(bn_l + 0), nW1T[l], 384, 128);
        T(F(bn_l + 1), nW2T[l], 128, 128);
        T(F(bn_l + 2), nW3T[l], 128, 128);
        T(F(bn_l + 3), nWd1T[l], 128, 512);
        T(F(bn_l + 4), nWd2T[l], 512, 128);
    }

    auto ggrid = [](int M, int Nd) { return dim3(M / 16, (((Nd + 15) / 16) + 3) / 4); };

    // ---- features ----
    knn_kernel<<<NV, 256, 0, stream>>>(Xin, x_mask, Eidx, Dnbr);
    dihedral_kernel<<<(NV * 3 + 127) / 128, 128, 0, stream>>>(Xin, angles);
    vfeat_kernel<<<NV, 128, 0, stream>>>(angles, F(9), F(14), F(58), F(102), V_embed, Ahv);
    gemm_f16_wmma<false, true, false><<<ggrid(NV, 128), 128, 0, stream>>>(
        Ahv, WvT, F(13), hV, nullptr, NV, 256, 128);
    eraw_kernel<<<NE, 128, 0, stream>>>(Eidx, chain, Dnbr, E_embed, Aeraw, Ahe);
    gemm_f16_wmma<false, true, false><<<ggrid(NE, 128), 128, 0, stream>>>(
        Aeraw, WefT, F(11), m3, nullptr, NE, 32, 128);
    fused_add_ln<<<NE, 128, 0, stream>>>(m3, nullptr, nullptr, F(57), F(59), nullptr,
                                         nullptr, Ahe, 256, 0);
    gemm_f16_wmma<false, true, false><<<ggrid(NE, 128), 128, 0, stream>>>(
        Ahe, WeT, F(10), hE_a, nullptr, NE, 256, 128);
    edge_masks<<<(NE + 255) / 256, 256, 0, stream>>>(Eidx, x_mask, mAtt, mE);
    reverse_edges<<<(NE + 255) / 256, 256, 0, stream>>>(Eidx, exst, kpr);
    merge128<<<NE, 128, 0, stream>>>(hE_a, hE_b, Eidx, exst, kpr);
    float* cur = hE_b;
    float* alt = hE_a;

    // ---- message-passing layers ----
    for (int l = 0; l < NLAY; ++l) {
        const int be_l = 15 + 14 * l, bn_l = 60 + 14 * l;
        // edge MPNN
        build_edge_x<<<NE, 512, 0, stream>>>(cur, hV, Eidx, R1);
        gemm_f16_wmma<true, false, true><<<ggrid(NE, 128), 128, 0, stream>>>(
            R1, eW1T[l], F(be_l + 5), nullptr, M1, NE, 512, 128);
        gemm_f16_wmma<true, false, true><<<ggrid(NE, 128), 128, 0, stream>>>(
            M1, eW2T[l], F(be_l + 6), nullptr, M2, NE, 128, 128);
        gemm_f16_wmma<false, true, false><<<ggrid(NE, 128), 128, 0, stream>>>(
            M2, eW3T[l], F(be_l + 7), m3, nullptr, NE, 128, 128);
        fused_add_ln<<<NE, 128, 0, stream>>>(cur, m3, mAtt, F(be_l + 10), F(be_l + 12),
                                             nullptr, hEln, hEln16, 128, 0);
        gemm_f16_wmma<true, false, true><<<ggrid(NE, 512), 128, 0, stream>>>(
            hEln16, eWd1T[l], F(be_l + 8), nullptr, R1, NE, 128, 512);
        gemm_f16_wmma<false, true, false><<<ggrid(NE, 128), 128, 0, stream>>>(
            R1, eWd2T[l], F(be_l + 9), m3, nullptr, NE, 512, 128);
        fused_add_ln<<<NE, 128, 0, stream>>>(hEln, m3, nullptr, F(be_l + 11), F(be_l + 13),
                                             mE, alt, nullptr, 128, 0);
        { float* t = cur; cur = alt; alt = t; }
        merge128<<<NE, 128, 0, stream>>>(cur, alt, Eidx, exst, kpr);
        { float* t = cur; cur = alt; alt = t; }
        // node MPNN
        build_node_x<<<NE, 384, 0, stream>>>(cur, hV, Eidx, R1);
        gemm_f16_wmma<true, false, true><<<ggrid(NE, 128), 128, 0, stream>>>(
            R1, nW1T[l], F(bn_l + 5), nullptr, M1, NE, 384, 128);
        gemm_f16_wmma<true, false, true><<<ggrid(NE, 128), 128, 0, stream>>>(
            M1, nW2T[l], F(bn_l + 6), nullptr, M2, NE, 128, 128);
        gemm_f16_wmma<false, true, false><<<ggrid(NE, 128), 128, 0, stream>>>(
            M2, nW3T[l], F(bn_l + 7), m3, nullptr, NE, 128, 128);
        node_reduce<<<(NV * HD + 255) / 256, 256, 0, stream>>>(m3, mAtt, hVdh);
        fused_add_ln<<<NV, 128, 0, stream>>>(hV, hVdh, nullptr, F(bn_l + 10), F(bn_l + 12),
                                             nullptr, hVln, hVln16, 128, 0);
        gemm_f16_wmma<true, false, true><<<ggrid(NV, 512), 128, 0, stream>>>(
            hVln16, nWd1T[l], F(bn_l + 8), nullptr, hVd116, NV, 128, 512);
        gemm_f16_wmma<false, true, false><<<ggrid(NV, 128), 128, 0, stream>>>(
            hVd116, nWd2T[l], F(bn_l + 9), hVdh, nullptr, NV, 512, 128);
        fused_add_ln<<<NV, 128, 0, stream>>>(hVln, hVdh, nullptr, F(bn_l + 11), F(bn_l + 13),
                                             x_mask, hV, nullptr, 128, 0);
    }

    // ---- output head ----
    float* outE = (float*)d_out;
    float* outV = outE + (size_t)NE * OD;
    int*   outI = (int*)(outV + (size_t)NV * HD);
    cvt_f32_f16<<<(NE * HD + 255) / 256, 256, 0, stream>>>(cur, hE16, NE * HD);
    gemm_f16_wmma<false, true, false><<<ggrid(NE, 400), 128, 0, stream>>>(
        hE16, WoutT, F(12), out400, nullptr, NE, 128, 400);
    selfslot_mask<<<NE, 400, 0, stream>>>(out400, mE);
    merge_pairE<<<NE, 400, 0, stream>>>(out400, Eidx, exst, kpr, outE);
    copy_f32k<<<(NV * HD + 255) / 256, 256, 0, stream>>>(hV, outV, NV * HD);
    copy_i32k<<<(NE + 255) / 256, 256, 0, stream>>>(Eidx, outI, NE);
}
